// Router_24103356465247
// MI455X (gfx1250) — matrix operand order repaired and checked by reading, not compiled, not run on hardware
//
#include <hip/hip_runtime.h>
#include <hip/hip_bf16.h>
#include <math.h>

// ---------------------------------------------------------------------------
// MoE router for MI455X (gfx1250, wave32).
//   logits = x[T,2048] @ W^T[2048,64] via bf16 split-precision WMMA:
//     x ~ x_hi+x_lo, W ~ w_hi+w_lo, logit = x_hi*w_hi + x_lo*w_hi + x_hi*w_lo
//   W_hi/W_lo bf16 planes precomputed in workspace. Per K=32 step each
//   workgroup async-DMAs the 8KB B slab into double-buffered LDS
//   (global_load_async_to_lds_b128 / ASYNCcnt) shared by all 4 waves, and
//   each wave pipelines its own non-temporal A loads one step ahead.
//   12 v_wmma_f32_16x16x32_bf16 per step per wave.
// ---------------------------------------------------------------------------

typedef __attribute__((ext_vector_type(16))) __bf16       v16bf;
typedef __attribute__((ext_vector_type(8)))  __bf16       v8bf;
typedef __attribute__((ext_vector_type(8)))  float        v8f;
typedef __attribute__((ext_vector_type(4)))  float        f32x4;
typedef __attribute__((ext_vector_type(4)))  unsigned int u32x4;
typedef __attribute__((ext_vector_type(4)))  int          v4i;

typedef __attribute__((address_space(1))) v4i* global_v4i_ptr;
typedef __attribute__((address_space(3))) v4i* lds_v4i_ptr;

#define D_DIM   2048
#define E_DIM   64
#define BSTRIDE 40   // bf16 elems per expert row in LDS (80B: conflict-free, 16B-aligned)

#if __has_builtin(__builtin_amdgcn_global_load_async_to_lds_b128)
#define HAVE_ASYNC_LDS 1
#else
#define HAVE_ASYNC_LDS 0
#endif

template <int N>
__device__ __forceinline__ void wait_async_le()
{
#if HAVE_ASYNC_LDS
#if __has_builtin(__builtin_amdgcn_s_wait_asynccnt)
    __builtin_amdgcn_s_wait_asynccnt(N);
#else
    asm volatile("s_wait_asynccnt %0" :: "n"(N) : "memory");
#endif
#endif
}

// Stage one K=32 slab of W_hi/W_lo (2 planes x 64 experts x 64B = 8KB) into LDS.
// 128 threads x 4 chunks of 16B each.
__device__ __forceinline__ void stage_w(const __bf16* __restrict__ whi,
                                        const __bf16* __restrict__ wlo,
                                        __bf16 (*dst)[E_DIM][BSTRIDE],
                                        int k0, int tid)
{
    #pragma unroll
    for (int t = 0; t < 4; ++t) {
        int c  = tid + t * 128;          // 0..511
        int pl = c >> 8;                 // 0 = hi plane, 1 = lo plane
        int cc = c & 255;
        int e  = cc >> 2;                // expert
        int q  = cc & 3;                 // 16B chunk within 64B K-row
        const __bf16* g = (pl ? wlo : whi) + (size_t)e * D_DIM + k0 + q * 8;
        __bf16* l = &dst[pl][e][q * 8];
#if HAVE_ASYNC_LDS
        __builtin_amdgcn_global_load_async_to_lds_b128(
            (global_v4i_ptr)g, (lds_v4i_ptr)l, 0, 0);
#else
        *(u32x4*)l = *(const u32x4*)g;   // sync fallback: load + ds_store
#endif
    }
}

__device__ __forceinline__ void compute_step(
    const __bf16 (&B)[2][E_DIM][BSTRIDE],
    const f32x4 A[4], int r, int h,
    v8f& acc0, v8f& acc1, v8f& acc2, v8f& acc3)
{
    float av[16];
    #pragma unroll
    for (int j = 0; j < 4; ++j) {
        av[j]      = A[0][j]; av[j + 4]  = A[1][j];
        av[j + 8]  = A[2][j]; av[j + 12] = A[3][j];
    }
    // fp32 -> bf16 hi/lo split of A (only per-element conversion in hot loop)
    v16bf ahi, alo;
    #pragma unroll
    for (int j = 0; j < 16; ++j) {
        float f = av[j];
        __bf16 hi = (__bf16)f;
        ahi[j] = hi;
        alo[j] = (__bf16)(f - (float)hi);
    }

#define ROUTER_TILE(n, ACC)                                                       \
    {                                                                             \
        const v8bf* ph = reinterpret_cast<const v8bf*>(&B[0][16 * (n) + r][8 * h]); \
        const v8bf* pl = reinterpret_cast<const v8bf*>(&B[1][16 * (n) + r][8 * h]); \
        v8bf bh0 = ph[0], bh1 = ph[2], bl0 = pl[0], bl1 = pl[2];                  \
        v16bf bh = __builtin_shufflevector(bh0, bh1, 0,1,2,3,4,5,6,7,8,9,10,11,12,13,14,15); \
        v16bf bl = __builtin_shufflevector(bl0, bl1, 0,1,2,3,4,5,6,7,8,9,10,11,12,13,14,15); \
        ACC = __builtin_amdgcn_wmma_f32_16x16x32_bf16(false, ahi, false, bh,      \
                                              (short)0, ACC, false, false);       \
        ACC = __builtin_amdgcn_wmma_f32_16x16x32_bf16(false, alo, false, bh,      \
                                              (short)0, ACC, false, false);       \
        ACC = __builtin_amdgcn_wmma_f32_16x16x32_bf16(false, ahi, false, bl,      \
                                              (short)0, ACC, false, false);       \
    }
    ROUTER_TILE(0, acc0)
    ROUTER_TILE(1, acc1)
    ROUTER_TILE(2, acc2)
    ROUTER_TILE(3, acc3)
#undef ROUTER_TILE
}

__global__ void __launch_bounds__(128)
router_gemm_kernel(const float* __restrict__ x,
                   const __bf16* __restrict__ whi,
                   const __bf16* __restrict__ wlo,
                   float* __restrict__ out_idx, float* __restrict__ out_w,
                   float* __restrict__ ws_usage, int T)
{
    __shared__ __bf16 ldsB[2][2][E_DIM][BSTRIDE];   // [buf][plane][expert][K] ~20.5KB
    __shared__ float  lds_logits[64][65];
    __shared__ float  lds_rmax[64];
    __shared__ float  lds_rinv[64];

    const int tid  = threadIdx.x;
    const int wave = tid >> 5;            // each wave: its own 16-row M-tile
    const int lane = tid & 31;
    const int r    = lane & 15;
    const int h    = lane >> 4;           // lane half (K-run selector)
    const int rowBase = blockIdx.x * 64;
    const int waveRow = rowBase + wave * 16;

    const float* __restrict__ xrow = x + (size_t)(waveRow + r) * D_DIM;
    // A 16x32 bf16 layout: lane<16 holds K{0..7,16..23}; lane>=16 K{8..15,24..31}
    const int aOff0 = h * 8;
    const int aOff1 = 16 + h * 8;

    v8f acc0 = {0.f,0.f,0.f,0.f,0.f,0.f,0.f,0.f};
    v8f acc1 = acc0, acc2 = acc0, acc3 = acc0;

    // ---- prologue: stage B slab 0, load A slab 0 ----
    stage_w(whi, wlo, ldsB[0], 0, tid);
    f32x4 A[4], An[4];
    {
        const f32x4* pa0 = reinterpret_cast<const f32x4*>(xrow + aOff0);
        const f32x4* pa1 = reinterpret_cast<const f32x4*>(xrow + aOff1);
        A[0] = __builtin_nontemporal_load(pa0);
        A[1] = __builtin_nontemporal_load(pa0 + 1);
        A[2] = __builtin_nontemporal_load(pa1);
        A[3] = __builtin_nontemporal_load(pa1 + 1);
    }
    int buf = 0;

    // ---- steady state: 63 pipelined steps ----
    for (int k0 = 0; k0 < D_DIM - 32; k0 += 32) {
        // prefetch next A slab (non-temporal; x is streamed exactly once)
        const f32x4* pa0 = reinterpret_cast<const f32x4*>(xrow + k0 + 32 + aOff0);
        const f32x4* pa1 = reinterpret_cast<const f32x4*>(xrow + k0 + 32 + aOff1);
        An[0] = __builtin_nontemporal_load(pa0);
        An[1] = __builtin_nontemporal_load(pa0 + 1);
        An[2] = __builtin_nontemporal_load(pa1);
        An[3] = __builtin_nontemporal_load(pa1 + 1);

        __syncthreads();                          // prior reads of fill buffer done
        stage_w(whi, wlo, ldsB[buf ^ 1], k0 + 32, tid);
        wait_async_le<4>();                       // older slab landed; prefetch stays in flight
        __syncthreads();                          // slab visible to all waves

        compute_step(ldsB[buf], A, r, h, acc0, acc1, acc2, acc3);

        A[0] = An[0]; A[1] = An[1]; A[2] = An[2]; A[3] = An[3];
        buf ^= 1;
    }
    // ---- final step (no further staging) ----
    wait_async_le<0>();
    __syncthreads();
    compute_step(ldsB[buf], A, r, h, acc0, acc1, acc2, acc3);

    // C layout: VGPR i -> (M = i + 8*h, N = 16*tile + r)
    #pragma unroll
    for (int i = 0; i < 8; ++i) {
        int row = wave * 16 + i + 8 * h;
        lds_logits[row][r]      = acc0[i];
        lds_logits[row][16 + r] = acc1[i];
        lds_logits[row][32 + r] = acc2[i];
        lds_logits[row][48 + r] = acc3[i];
    }
    __syncthreads();

    // --- per-row softmax / top-2 (one thread per row, 64 rows) ---
    if (tid < 64) {
        float v1 = -INFINITY, v2 = -INFINITY;
        int   e1 = 0, e2 = 0;
        #pragma unroll 4
        for (int e = 0; e < E_DIM; ++e) {
            float v = lds_logits[tid][e];
            if (v > v1)      { v2 = v1; e2 = e1; v1 = v; e1 = e; }
            else if (v > v2) { v2 = v;  e2 = e; }
        }
        float mx = v1;
        float s = 0.f;
        #pragma unroll 4
        for (int e = 0; e < E_DIM; ++e)
            s += __expf(lds_logits[tid][e] - mx);

        float x1 = __expf(v1 - mx);               // == 1
        float x2 = __expf(v2 - mx);
        float inv12 = 1.f / (x1 + x2);            // top-2 renorm (full softmax cancels)
        size_t t = (size_t)rowBase + tid;
        out_idx[t * 2 + 0] = (float)e1;
        out_idx[t * 2 + 1] = (float)e2;
        out_w[t * 2 + 0]   = x1 * inv12;
        out_w[t * 2 + 1]   = x2 * inv12;
        lds_rmax[tid] = mx;
        lds_rinv[tid] = 1.f / s;
    }
    __syncthreads();

    // --- per-expert usage partial sums (one thread per expert, 64 rows) ---
    if (tid < E_DIM) {
        float s = 0.f;
        #pragma unroll 4
        for (int rr = 0; rr < 64; ++rr)
            s += __expf(lds_logits[rr][tid] - lds_rmax[rr]) * lds_rinv[rr];
        atomicAdd(ws_usage + tid, s);
    }
}

// Pre-split W (E*D fp32, 512KB) into bf16 hi/lo planes once per launch.
__global__ void __launch_bounds__(256)
prep_w_kernel(const float* __restrict__ W,
              __bf16* __restrict__ whi, __bf16* __restrict__ wlo)
{
    int i = blockIdx.x * 256 + threadIdx.x;
    float f = W[i];
    __bf16 hi = (__bf16)f;
    whi[i] = hi;
    wlo[i] = (__bf16)(f - (float)hi);
}

__global__ void zero_ws_kernel(float* __restrict__ ws)
{
    ws[threadIdx.x] = 0.f;
}

__global__ void aux_loss_kernel(const float* __restrict__ ws_usage,
                                float* __restrict__ out_aux, int T)
{
    __shared__ float red[E_DIM];
    int e = threadIdx.x;
    float u = ws_usage[e] / (float)T;
    float d = u - (1.0f / (float)E_DIM);
    red[e] = d * d;
    __syncthreads();
    #pragma unroll
    for (int s = E_DIM / 2; s > 0; s >>= 1) {
        if (e < s) red[e] += red[e + s];
        __syncthreads();
    }
    if (e == 0) *out_aux = red[0];
}

extern "C" void kernel_launch(void* const* d_in, const int* in_sizes, int n_in,
                              void* d_out, int out_size, void* d_ws, size_t ws_size,
                              hipStream_t stream)
{
    const float* x = (const float*)d_in[0];   // [B,S,D] fp32
    const float* W = (const float*)d_in[1];   // [E,D]   fp32

    const int T = in_sizes[0] / D_DIM;        // B*S = 16384

    float* out      = (float*)d_out;
    float* out_idx  = out;                    // T*2 indices (as float values)
    float* out_w    = out + (size_t)T * 2;    // T*2 weights
    float* out_aux  = out + (size_t)T * 4;    // scalar aux loss

    // workspace: [64 floats usage][E*D bf16 W_hi][E*D bf16 W_lo] (~512.5KB)
    float*  ws_usage = (float*)d_ws;
    __bf16* whi = (__bf16*)((char*)d_ws + 256);
    __bf16* wlo = (__bf16*)((char*)d_ws + 256 + (size_t)E_DIM * D_DIM * 2);

    zero_ws_kernel<<<1, E_DIM, 0, stream>>>(ws_usage);
    prep_w_kernel<<<(E_DIM * D_DIM) / 256, 256, 0, stream>>>(W, whi, wlo);
    router_gemm_kernel<<<T / 64, 128, 0, stream>>>(x, whi, wlo, out_idx, out_w,
                                                   ws_usage, T);
    aux_loss_kernel<<<1, E_DIM, 0, stream>>>(ws_usage, out_aux, T);
}
